// ESN_300647710854
// MI455X (gfx1250) — matrix operand order repaired and checked by reading, not compiled, not run on hardware
//
#include <hip/hip_runtime.h>
#include <hip/hip_bf16.h>
#include <math.h>

// ESN recurrent reservoir on gfx1250 (MI455X), bf16 WMMA + persistent grid-sync kernel.
//
//   h_t = tanh([x_t | h_{t-1}] @ Wcat^T),  Wcat = [Win | W] (bf16, K padded 300->320)
//   t>=2: h_t = 0.5*tanh(...) + 0.5*h_{t-1}
//
// One persistent kernel runs all 512 steps with a device-wide atomic barrier per step.
// 512 waves == 512 output tiles (16x16) per step; each wave: 74 v_wmma_f32_16x16x32_bf16.
// Epilogue uses the gfx1250 hardware V_TANH_F32 transcendental instead of OCML tanhf.

typedef __attribute__((ext_vector_type(16))) __bf16 v16bf;
typedef __attribute__((ext_vector_type(8)))  float  v8f;

#define B_    64
#define T_    512
#define D_    300
#define R_    2048
#define KX    320              // x-columns padded to multiple of 32
#define KP    (KX + R_)        // 2368 total K per step
#define KT    32               // WMMA K
#define NKI   (KP / KT)        // 74
#define NWG   64
#define TPB   256
#define LEAKY 0.5f

union Frag16 { v16bf v; uint4 q[2]; };

// Hardware tanh (V_TANH_F32) on gfx1250; single TRANS op vs ~40 VALU ops for OCML tanhf.
__device__ __forceinline__ float esn_tanh(float v) {
#if __has_builtin(__builtin_amdgcn_tanhf)
    return __builtin_amdgcn_tanhf(v);
#else
    return tanhf(v);
#endif
}

// ---- Build packed bf16 weights: Wcat[n][k] = k<300 ? Win[n][k] : (k<320 ? 0 : W[n][k-320])
__global__ void esn_pack_weights(const float* __restrict__ Win,
                                 const float* __restrict__ W,
                                 __bf16* __restrict__ Wcat) {
    size_t idx = (size_t)blockIdx.x * blockDim.x + threadIdx.x;
    if (idx >= (size_t)R_ * KP) return;
    int n = (int)(idx / KP);
    int k = (int)(idx % KP);
    float v = 0.f;
    if (k < D_)       v = Win[(size_t)n * D_ + k];
    else if (k >= KX) v = W[(size_t)n * R_ + (k - KX)];
    Wcat[idx] = (__bf16)v;
}

// ---- Init state buffer for t=0: [x_0 | zeros(pad) | h_{-1}=0]
__global__ void esn_init_state(const float* __restrict__ x, __bf16* __restrict__ S0) {
    int idx = blockIdx.x * blockDim.x + threadIdx.x;
    if (idx >= B_ * KP) return;
    int b = idx / KP, k = idx - b * KP;
    float v = (k < D_) ? x[(size_t)b * T_ * D_ + k] : 0.f;   // x[b][t=0][k]
    S0[idx] = (__bf16)v;
}

// ---- Persistent step kernel: 64 WGs x 8 waves; wave gw owns output tile (m0,n0)
__global__ void __launch_bounds__(TPB)
esn_steps(const float*  __restrict__ x,
          const __bf16* __restrict__ Wcat,
          __bf16* __restrict__ S0,
          __bf16* __restrict__ S1,
          float*  __restrict__ out,
          unsigned int* __restrict__ cnt) {
    const int tid  = threadIdx.x;
    const int lane = tid & 31;
    const int wid  = tid >> 5;
    const int gw   = blockIdx.x * (TPB / 32) + wid;   // 0..511
    const int m0   = (gw & 3) * 16;                   // batch tile (64/16 = 4)
    const int n0   = (gw >> 2) * 16;                  // reservoir tile (2048/16 = 128)

    const int lmod = lane & 15;
    const int lhi  = lane >> 4;

    // A fragment (16x32 bf16, ISA layout): lane holds row m0+lmod;
    //   chunk0 = 16B at k0 + 8*lhi, chunk1 = 16B at k0 + 16 + 8*lhi
    const size_t aRow = (size_t)(m0 + lmod) * KP + 8 * lhi;
    // B fragment (32x16 bf16): lane holds col n0+lmod; chunk0 = 16B at k0 + 8*lhi, chunk1 = 16B at k0 + 16 + 8*lhi
    const __bf16* const bRow = Wcat + (size_t)(n0 + lmod) * KP + 8 * lhi;

    float hprev[8];                                   // this wave's own tile from step t-1
    #pragma unroll
    for (int j = 0; j < 8; ++j) hprev[j] = 0.f;

    for (int t = 0; t < T_; ++t) {
        const __bf16* Scur = (t & 1) ? S1 : S0;
        __bf16*       Snxt = (t & 1) ? S0 : S1;

        v8f c = {};
        const __bf16* aPtr = Scur + aRow;
        const __bf16* bPtr = bRow;
        #pragma unroll 2
        for (int ki = 0; ki < NKI; ++ki) {
            Frag16 a, b;
            a.q[0] = *(const uint4*)(aPtr);
            a.q[1] = *(const uint4*)(aPtr + 16);
            b.q[0] = *(const uint4*)(bPtr);
            b.q[1] = *(const uint4*)(bPtr + 16);
            __builtin_prefetch(bPtr + 4 * KT, 0, 1);  // stream W from L2 (global_prefetch_b8)
            c = __builtin_amdgcn_wmma_f32_16x16x32_bf16(false, a.v, false, b.v,
                                                        (short)0, c, false, false);
            aPtr += KT;
            bPtr += KT;
        }

        // Epilogue: C layout -> lane lmod = N, VGPR j -> M = j + 8*lhi
        float* outT = out + (size_t)t * (B_ * R_);
        #pragma unroll
        for (int j = 0; j < 8; ++j) {
            const int m = m0 + j + 8 * lhi;
            const int n = n0 + lmod;
            float h = esn_tanh(c[j]);                 // v_tanh_f32 (hardware TRANS op)
            if (t >= 2) h = (1.0f - LEAKY) * h + LEAKY * hprev[j];
            hprev[j] = h;
            outT[(size_t)m * R_ + n] = h;                         // f32 result
            Snxt[(size_t)m * KP + KX + n] = (__bf16)h;            // bf16 state for t+1
        }

        // Stage x_{t+1} (f32 -> bf16, zero-pad 300..319) into next state buffer
        if (t + 1 < T_) {
            for (int idx = blockIdx.x * TPB + tid; idx < B_ * KX; idx += NWG * TPB) {
                int b = idx / KX, d = idx - b * KX;
                float v = (d < D_) ? x[((size_t)b * T_ + (t + 1)) * D_ + d] : 0.f;
                Snxt[(size_t)b * KP + d] = (__bf16)v;
            }
        }

        // ---- Device-wide barrier (monotonic counter; counter zeroed per launch) ----
        __threadfence();
        __syncthreads();
        if (tid == 0) {
            __hip_atomic_fetch_add(cnt, 1u, __ATOMIC_RELEASE, __HIP_MEMORY_SCOPE_AGENT);
            const unsigned target = (unsigned)(NWG * (t + 1));
            while (__hip_atomic_load(cnt, __ATOMIC_ACQUIRE, __HIP_MEMORY_SCOPE_AGENT) < target)
                __builtin_amdgcn_s_sleep(2);
        }
        __syncthreads();
        __threadfence();
    }
}

extern "C" void kernel_launch(void* const* d_in, const int* in_sizes, int n_in,
                              void* d_out, int out_size, void* d_ws, size_t ws_size,
                              hipStream_t stream) {
    (void)in_sizes; (void)n_in; (void)out_size; (void)ws_size;
    const float* x   = (const float*)d_in[0];   // [64, 512, 300]
    const float* Win = (const float*)d_in[1];   // [2048, 300]
    const float* W   = (const float*)d_in[2];   // [2048, 2048]
    float* out = (float*)d_out;                 // [512, 64, 2048]

    // Workspace layout (~10 MB): [counter 256B][Wcat bf16 9.25MB][S0][S1]
    char* ws = (char*)d_ws;
    unsigned int* cnt = (unsigned int*)ws;
    __bf16* Wcat = (__bf16*)(ws + 256);
    size_t wcatBytes = (size_t)R_ * KP * sizeof(__bf16);        // 256B-aligned size
    __bf16* S0 = (__bf16*)(ws + 256 + ((wcatBytes + 255) & ~(size_t)255));
    __bf16* S1 = S0 + (size_t)B_ * KP;                          // 303104B, 256B-aligned

    hipMemsetAsync(cnt, 0, 256, stream);

    {
        size_t total = (size_t)R_ * KP;
        unsigned blocks = (unsigned)((total + 255) / 256);
        esn_pack_weights<<<blocks, 256, 0, stream>>>(Win, W, Wcat);
    }
    esn_init_state<<<(B_ * KP + 255) / 256, 256, 0, stream>>>(x, S0);
    esn_steps<<<NWG, TPB, 0, stream>>>(x, Wcat, S0, S1, out, cnt);
}
